// MultiheadSelfAttentionBlock_57354993271087
// MI455X (gfx1250) — matrix operand order repaired and checked by reading, not compiled, not run on hardware
//
#include <hip/hip_runtime.h>
#include <hip/hip_bf16.h>

// MI455X (gfx1250) CDNA5 implementation: bf16 WMMA for all four GEMMs
// (f32 accumulate), fully-unrolled software-pipelined fragment loads,
// TDM (tensor_load_to_lds) staging of Q-tiles into LDS, LDS-tiled masked
// softmax with per-head attn-weight accumulation, fused residual+LayerNorm.

#define B_ 8
#define S_ 1024
#define E_ 1024
#define H_ 16
#define DH_ 64

typedef __bf16 bf16_t;
typedef __attribute__((ext_vector_type(16))) __bf16 bf16x16;
typedef __attribute__((ext_vector_type(8)))  __bf16 bf16x8;
typedef __attribute__((ext_vector_type(8)))  float  f32x8;
typedef __attribute__((ext_vector_type(4)))  unsigned int u32x4;
typedef __attribute__((ext_vector_type(4)))  int    i32x4;
typedef __attribute__((ext_vector_type(8)))  int    i32x8;

union ABfrag { bf16x16 v; bf16x8 h[2]; };

// ---------------------------------------------------------------------------
// K1: mask src -> f32 residual + bf16 activations
// ---------------------------------------------------------------------------
__global__ __launch_bounds__(256) void prep_kernel(
    const float* __restrict__ src, const unsigned char* __restrict__ pm,
    float* __restrict__ x_res, bf16_t* __restrict__ xb) {
  const int row = blockIdx.x;            // B*S rows
  const bool masked = pm[row] != 0;
  const float* s = src + (size_t)row * E_;
  float* xr = x_res + (size_t)row * E_;
  bf16_t* xo = xb + (size_t)row * E_;
  for (int e = threadIdx.x; e < E_; e += 256) {
    float v = masked ? 0.0f : s[e];
    xr[e] = v;
    xo[e] = (bf16_t)v;
  }
}

// ---------------------------------------------------------------------------
// K2: f32 -> bf16 weight conversion
// ---------------------------------------------------------------------------
__global__ __launch_bounds__(256) void cvt_bf16_kernel(
    const float* __restrict__ w, bf16_t* __restrict__ o, int n) {
  int i = blockIdx.x * 256 + threadIdx.x;
  if (i < n) o[i] = (bf16_t)w[i];
}

// ---------------------------------------------------------------------------
// K3: packed QKV GEMM.  Y[M=8192, N=3072] = X[M,1024] * W[N,1024]^T + bias.
// Block = 8 waves -> 64(M) x 128(N) macrotile; wave = 16x64 (4 accumulators).
// k-loop FULLY UNROLLED with next-step fragments loaded before current WMMAs:
// ping-pong buffers become SSA values (no register copies), addresses fold to
// load immediates, and the scheduler overlaps L2 latency with WMMA issue.
// Output split: n<E -> Q[B,S,E] bf16 ; n<2E -> K[B,S,E] bf16 ;
//               else -> V transposed vT[B,H,DH,S] bf16 (contiguous store).
// ---------------------------------------------------------------------------
__global__ __launch_bounds__(256) void qkv_gemm_kernel(
    const bf16_t* __restrict__ X, const bf16_t* __restrict__ W,
    const float* __restrict__ bias,
    bf16_t* __restrict__ qb, bf16_t* __restrict__ kb, bf16_t* __restrict__ vT) {
  const int Kd = E_;
  const int lane = threadIdx.x & 31;
  const int wave = threadIdx.x >> 5;
  const int wm = wave & 3, wn = wave >> 2;
  const int m0 = blockIdx.y * 64 + wm * 16;
  const int n0 = blockIdx.x * 128 + wn * 64;
  const int half = lane >> 4, l16 = lane & 15;

  f32x8 acc[4] = {{}, {}, {}, {}};

  const bf16_t* aRow = X + (size_t)(m0 + l16) * Kd + half * 16;
  const bf16_t* bCol = W + (size_t)(n0 + l16) * Kd + half * 16;

  ABfrag a;
  a.h[0] = *(const bf16x8*)(aRow);
  a.h[1] = *(const bf16x8*)(aRow + 8);
  bf16x16 bfr[4];
#pragma unroll
  for (int nb = 0; nb < 4; ++nb)
    bfr[nb] = *(const bf16x16*)(bCol + (size_t)nb * 16 * Kd);

#pragma unroll
  for (int kk = 0; kk < Kd; kk += 32) {
    const bool more = (kk + 32) < Kd;   // compile-time after full unroll
    ABfrag an;
    bf16x16 bn[4];
    if (more) {
      __builtin_prefetch(aRow + kk + 512, 0, 1);
      an.h[0] = *(const bf16x8*)(aRow + kk + 32);
      an.h[1] = *(const bf16x8*)(aRow + kk + 40);
#pragma unroll
      for (int nb = 0; nb < 4; ++nb)
        bn[nb] = *(const bf16x16*)(bCol + (size_t)nb * 16 * Kd + kk + 32);
    }
#pragma unroll
    for (int nb = 0; nb < 4; ++nb)
      acc[nb] = __builtin_amdgcn_wmma_f32_16x16x32_bf16(
          false, a.v, false, bfr[nb], (short)0, acc[nb], false, false);
    if (more) {
      a = an;
#pragma unroll
      for (int nb = 0; nb < 4; ++nb) bfr[nb] = bn[nb];
    }
  }

  const int mbase = m0 + half * 8;
#pragma unroll
  for (int nb = 0; nb < 4; ++nb) {
    const int n = n0 + nb * 16 + l16;
    const float bn = bias[n];
    const f32x8 av = acc[nb];
    if (n < E_) {
#pragma unroll
      for (int r = 0; r < 8; ++r)
        qb[(size_t)(mbase + r) * E_ + n] = (bf16_t)(av[r] + bn);
    } else if (n < 2 * E_) {
#pragma unroll
      for (int r = 0; r < 8; ++r)
        kb[(size_t)(mbase + r) * E_ + (n - E_)] = (bf16_t)(av[r] + bn);
    } else {
      const int nn = n - 2 * E_;
      const int hh = nn >> 6, d = nn & 63;
      const int bidx = mbase >> 10;       // m tiles are 64-aligned, never cross batch
      const int s0 = mbase & 1023;
      bf16x8 vs;
#pragma unroll
      for (int r = 0; r < 8; ++r) vs[r] = (bf16_t)(av[r] + bn);
      *(bf16x8*)(vT + ((size_t)((bidx * H_ + hh) * DH_ + d)) * S_ + s0) = vs;
    }
  }
}

// ---------------------------------------------------------------------------
// K4: attention.  One block per (b, 16-row q-tile); loops all heads so the
// head-mean attn_weights accumulates in LDS (single global write at the end).
// Q-tile (16x64 bf16) staged into LDS once per head via the Tensor Data Mover.
// LDS: sc[16][1032], aw[16][1032], red[256], ctxp[2][16][64], qtile[16][64]bf16
// ---------------------------------------------------------------------------
#define SCP 1032
__global__ __launch_bounds__(256) void attn_kernel(
    const bf16_t* __restrict__ qb, const bf16_t* __restrict__ kb,
    const bf16_t* __restrict__ vT, const unsigned char* __restrict__ pm,
    bf16_t* __restrict__ ctx, float* __restrict__ attn_w) {
  extern __shared__ float smem[];
  float*  sc    = smem;                     // 16*SCP
  float*  aw    = sc + 16 * SCP;            // 16*SCP
  float*  red   = aw + 16 * SCP;            // 256
  float*  ctxp  = red + 256;                // 2*16*64
  bf16_t* qtile = (bf16_t*)(ctxp + 2 * 16 * 64);  // 16*64 bf16 (2 KB)

  const int b = blockIdx.y;
  const int qt = blockIdx.x;
  const int tid = threadIdx.x;
  const int lane = tid & 31, wave = tid >> 5;
  const int half = lane >> 4, l16 = lane & 15;
  const int row_q0 = qt * 16;
  const float scale = 0.125f;               // 1/sqrt(DH)

  for (int i = tid; i < 16 * SCP; i += 256) aw[i] = 0.0f;
  __syncthreads();

  for (int h = 0; h < H_; ++h) {
    // ---- TDM: DMA the 16x64 Q tile for this head into LDS (wave 0 only) ---
    if (tid < 32) {
      const unsigned long long gaddr =
          (unsigned long long)(uintptr_t)(qb + (size_t)(b * S_ + row_q0) * E_ + h * DH_);
      const unsigned ldsaddr = (unsigned)(uintptr_t)qtile;
      // D# group0: count=1, lds_addr, 57-bit global_addr, type=2 ("image")
      u32x4 g0 = { 1u, ldsaddr,
                   (unsigned)(gaddr & 0xffffffffu),
                   (unsigned)((gaddr >> 32) & 0x01ffffffu) | 0x80000000u };
      // D# group1: data_size=2B; tensor_dim0=64, tensor_dim1=16;
      //            tile_dim0=64, tile_dim1=16; tensor_dim0_stride=E
      i32x8 g1 = { 0x00010000, (64 << 16), (16 << 16), (64 << 16),
                   16, E_, 0, 0 };
      i32x4 z4 = { 0, 0, 0, 0 };
#if __clang_major__ >= 23
      i32x8 z8 = { 0, 0, 0, 0, 0, 0, 0, 0 };
      __builtin_amdgcn_tensor_load_to_lds(g0, g1, z4, z4, z8, 0);
#else
      __builtin_amdgcn_tensor_load_to_lds(g0, g1, z4, z4, 0);
#endif
      __builtin_amdgcn_s_wait_tensorcnt((short)0);
    }
    __syncthreads();

    // ---- phase 1: scores = scale * Q Kᵀ (A from LDS), masked, into LDS ----
    const bf16_t* qrow = qtile + l16 * DH_ + half * 16;
    ABfrag a0, a1;
    a0.h[0] = *(const bf16x8*)(qrow);
    a0.h[1] = *(const bf16x8*)(qrow + 8);
    a1.h[0] = *(const bf16x8*)(qrow + 32);
    a1.h[1] = *(const bf16x8*)(qrow + 40);

    const bf16_t* kbase = kb + (size_t)(b * S_) * E_ + h * DH_ + half * 16;
    bf16x16 b0 = *(const bf16x16*)(kbase + (size_t)(wave * 128 + l16) * E_);
    bf16x16 b1 = *(const bf16x16*)(kbase + (size_t)(wave * 128 + l16) * E_ + 32);
#pragma unroll
    for (int i = 0; i < 8; ++i) {
      bf16x16 nb0, nb1;
      if (i < 7) {
        const bf16_t* np = kbase + (size_t)(wave * 128 + (i + 1) * 16 + l16) * E_;
        nb0 = *(const bf16x16*)np;
        nb1 = *(const bf16x16*)(np + 32);
      }
      f32x8 accs = {};
      accs = __builtin_amdgcn_wmma_f32_16x16x32_bf16(false, a0.v, false, b0, (short)0, accs, false, false);
      accs = __builtin_amdgcn_wmma_f32_16x16x32_bf16(false, a1.v, false, b1, (short)0, accs, false, false);
      const int skcol = wave * 128 + i * 16 + l16;
      const bool km = pm[b * S_ + skcol] != 0;
#pragma unroll
      for (int r = 0; r < 8; ++r)
        sc[(r + half * 8) * SCP + skcol] = km ? -__builtin_inff() : accs[r] * scale;
      if (i < 7) { b0 = nb0; b1 = nb1; }
    }
    __syncthreads();

    // ---- phase 2: masked softmax + attn-weight mean accumulation ----------
    {
      const int row = tid >> 4, sub = tid & 15;
      float lmax = -__builtin_inff();
      for (int j = 0; j < 64; ++j)
        lmax = fmaxf(lmax, sc[row * SCP + sub + j * 16]);
      red[row * 16 + sub] = lmax;
      __syncthreads();
      float rmax = -__builtin_inff();
      for (int j = 0; j < 16; ++j) rmax = fmaxf(rmax, red[row * 16 + j]);
      __syncthreads();
      float lsum = 0.0f;
      for (int j = 0; j < 64; ++j) {
        const int c = sub + j * 16;
        const float p = __expf(sc[row * SCP + c] - rmax);
        sc[row * SCP + c] = p;
        lsum += p;
      }
      red[row * 16 + sub] = lsum;
      __syncthreads();
      float rsum = 0.0f;
      for (int j = 0; j < 16; ++j) rsum += red[row * 16 + j];
      const float inv = 1.0f / rsum;
      for (int j = 0; j < 64; ++j) {
        const int c = sub + j * 16;
        const float p = sc[row * SCP + c] * inv;
        sc[row * SCP + c] = p;
        aw[row * SCP + c] += p * (1.0f / H_);
      }
    }
    __syncthreads();

    // ---- phase 3: ctx = P V  (waves split: 4 d-tiles x 2 k-halves) --------
    {
      const int dt = wave & 3;
      const int kh = wave >> 2;
      f32x8 accc = {};
      const bf16_t* vbase = vT + (size_t)((b * H_ + h) * DH_ + dt * 16 + l16) * S_;
#pragma unroll 4
      for (int j = kh * 16; j < kh * 16 + 16; ++j) {
        ABfrag a;
        const int kb0 = j * 32 + half * 16;
#pragma unroll
        for (int e = 0; e < 8; ++e) {
          a.v[e]     = (bf16_t)sc[l16 * SCP + kb0 + e];
          a.v[e + 8] = (bf16_t)sc[l16 * SCP + kb0 + 8 + e];
        }
        bf16x16 bv = *(const bf16x16*)(vbase + j * 32 + half * 16);
        accc = __builtin_amdgcn_wmma_f32_16x16x32_bf16(false, a.v, false, bv, (short)0, accc, false, false);
      }
      const int d = dt * 16 + l16;
#pragma unroll
      for (int r = 0; r < 8; ++r)
        ctxp[(kh * 16 + (r + half * 8)) * 64 + d] = accc[r];
    }
    __syncthreads();
    for (int i = tid; i < 16 * 64; i += 256) {
      const int m = i >> 6, d = i & 63;
      const float v = ctxp[m * 64 + d] + ctxp[(16 + m) * 64 + d];
      ctx[(size_t)(b * S_ + row_q0 + m) * E_ + h * DH_ + d] = (bf16_t)v;
    }
    __syncthreads();
  }

  for (int i = tid; i < 16 * 1024; i += 256) {
    const int m = i >> 10, c = i & 1023;
    attn_w[(size_t)(b * S_ + row_q0 + m) * S_ + c] = aw[m * SCP + c];
  }
}

// ---------------------------------------------------------------------------
// K5: out projection.  AO[M=8192, N=1024] = CTX[M,1024] * Wout[N,1024]^T + bias
// Same fully-unrolled pipelined structure as K3, f32 epilogue.
// ---------------------------------------------------------------------------
__global__ __launch_bounds__(256) void out_gemm_kernel(
    const bf16_t* __restrict__ X, const bf16_t* __restrict__ W,
    const float* __restrict__ bias, float* __restrict__ AO) {
  const int Kd = E_;
  const int lane = threadIdx.x & 31;
  const int wave = threadIdx.x >> 5;
  const int wm = wave & 3, wn = wave >> 2;
  const int m0 = blockIdx.y * 64 + wm * 16;
  const int n0 = blockIdx.x * 128 + wn * 64;
  const int half = lane >> 4, l16 = lane & 15;

  f32x8 acc[4] = {{}, {}, {}, {}};

  const bf16_t* aRow = X + (size_t)(m0 + l16) * Kd + half * 16;
  const bf16_t* bCol = W + (size_t)(n0 + l16) * Kd + half * 16;

  ABfrag a;
  a.h[0] = *(const bf16x8*)(aRow);
  a.h[1] = *(const bf16x8*)(aRow + 8);
  bf16x16 bfr[4];
#pragma unroll
  for (int nb = 0; nb < 4; ++nb)
    bfr[nb] = *(const bf16x16*)(bCol + (size_t)nb * 16 * Kd);

#pragma unroll
  for (int kk = 0; kk < Kd; kk += 32) {
    const bool more = (kk + 32) < Kd;   // compile-time after full unroll
    ABfrag an;
    bf16x16 bn[4];
    if (more) {
      __builtin_prefetch(aRow + kk + 512, 0, 1);
      an.h[0] = *(const bf16x8*)(aRow + kk + 32);
      an.h[1] = *(const bf16x8*)(aRow + kk + 40);
#pragma unroll
      for (int nb = 0; nb < 4; ++nb)
        bn[nb] = *(const bf16x16*)(bCol + (size_t)nb * 16 * Kd + kk + 32);
    }
#pragma unroll
    for (int nb = 0; nb < 4; ++nb)
      acc[nb] = __builtin_amdgcn_wmma_f32_16x16x32_bf16(
          false, a.v, false, bfr[nb], (short)0, acc[nb], false, false);
    if (more) {
      a = an;
#pragma unroll
      for (int nb = 0; nb < 4; ++nb) bfr[nb] = bn[nb];
    }
  }

  const int mbase = m0 + half * 8;
#pragma unroll
  for (int nb = 0; nb < 4; ++nb) {
    const int n = n0 + nb * 16 + l16;
    const float bn = bias[n];
    const f32x8 av = acc[nb];
#pragma unroll
    for (int r = 0; r < 8; ++r)
      AO[(size_t)(mbase + r) * E_ + n] = av[r] + bn;
  }
}

// ---------------------------------------------------------------------------
// K6: out = mask( LN( mask(attn_out) + residual ) )
// ---------------------------------------------------------------------------
__global__ __launch_bounds__(256) void ln_kernel(
    const float* __restrict__ ao, const float* __restrict__ x_res,
    const unsigned char* __restrict__ pm,
    const float* __restrict__ lnw, const float* __restrict__ lnb,
    float* __restrict__ out) {
  __shared__ float s1[256], s2[256];
  const int row = blockIdx.x;
  const bool masked = pm[row] != 0;
  const float* a = ao + (size_t)row * E_;
  const float* r = x_res + (size_t)row * E_;
  float t[4];
  float lsum = 0.0f, lsq = 0.0f;
#pragma unroll
  for (int i = 0; i < 4; ++i) {
    const int c = threadIdx.x + i * 256;
    const float v = (masked ? 0.0f : a[c]) + r[c];
    t[i] = v; lsum += v; lsq += v * v;
  }
  s1[threadIdx.x] = lsum; s2[threadIdx.x] = lsq;
  __syncthreads();
  for (int off = 128; off > 0; off >>= 1) {
    if (threadIdx.x < off) {
      s1[threadIdx.x] += s1[threadIdx.x + off];
      s2[threadIdx.x] += s2[threadIdx.x + off];
    }
    __syncthreads();
  }
  const float mu = s1[0] * (1.0f / E_);
  const float var = s2[0] * (1.0f / E_) - mu * mu;
  const float rstd = rsqrtf(var + 1e-5f);
#pragma unroll
  for (int i = 0; i < 4; ++i) {
    const int c = threadIdx.x + i * 256;
    const float y = (t[i] - mu) * rstd * lnw[c] + lnb[c];
    out[(size_t)row * E_ + c] = masked ? 0.0f : y;
  }
}

// ---------------------------------------------------------------------------
extern "C" void kernel_launch(void* const* d_in, const int* in_sizes, int n_in,
                              void* d_out, int out_size, void* d_ws, size_t ws_size,
                              hipStream_t stream) {
  (void)in_sizes; (void)n_in; (void)out_size; (void)ws_size;

  const float* src            = (const float*)d_in[0];
  const unsigned char* pm     = (const unsigned char*)d_in[1];  // bool mask [B,S]
  const float* in_proj_w      = (const float*)d_in[2];
  const float* in_proj_b      = (const float*)d_in[3];
  const float* out_proj_w     = (const float*)d_in[4];
  const float* out_proj_b     = (const float*)d_in[5];
  const float* ln_w           = (const float*)d_in[6];
  const float* ln_b           = (const float*)d_in[7];

  float* out_main = (float*)d_out;                       // [B,S,E]
  float* out_attn = out_main + (size_t)B_ * S_ * E_;     // [B,S,S]

  char* ws = (char*)d_ws;
  size_t off = 0;
  auto alloc = [&](size_t bytes) -> char* {
    char* p = ws + off;
    off += (bytes + 255) & ~(size_t)255;
    return p;
  };
  float*  x_res   = (float*) alloc((size_t)B_ * S_ * E_ * 4);
  bf16_t* xb      = (bf16_t*)alloc((size_t)B_ * S_ * E_ * 2);
  bf16_t* wqkvb   = (bf16_t*)alloc((size_t)3 * E_ * E_ * 2);
  bf16_t* woutb   = (bf16_t*)alloc((size_t)E_ * E_ * 2);
  bf16_t* qb      = (bf16_t*)alloc((size_t)B_ * S_ * E_ * 2);
  bf16_t* kb      = (bf16_t*)alloc((size_t)B_ * S_ * E_ * 2);
  bf16_t* vT      = (bf16_t*)alloc((size_t)B_ * S_ * E_ * 2);  // [B,H,DH,S]
  bf16_t* ctx     = (bf16_t*)alloc((size_t)B_ * S_ * E_ * 2);
  float*  attn_ao = (float*) alloc((size_t)B_ * S_ * E_ * 4);

  prep_kernel<<<B_ * S_, 256, 0, stream>>>(src, pm, x_res, xb);
  cvt_bf16_kernel<<<(3 * E_ * E_) / 256, 256, 0, stream>>>(in_proj_w, wqkvb, 3 * E_ * E_);
  cvt_bf16_kernel<<<(E_ * E_) / 256, 256, 0, stream>>>(out_proj_w, woutb, E_ * E_);

  qkv_gemm_kernel<<<dim3(24, 128), 256, 0, stream>>>(xb, wqkvb, in_proj_b, qb, kb, vT);

  const size_t smem =
      (size_t)(16 * SCP * 2 + 256 + 2 * 16 * 64 + 512) * sizeof(float); // ~140 KB + Q tile
  attn_kernel<<<dim3(S_ / 16, B_), 256, smem, stream>>>(qb, kb, vT, pm, ctx, out_attn);

  out_gemm_kernel<<<dim3(8, 128), 256, 0, stream>>>(ctx, woutb, out_proj_b, attn_ao);

  ln_kernel<<<B_ * S_, 256, 0, stream>>>(attn_ao, x_res, pm, ln_w, ln_b, out_main);
}
